// CustomGRU_72438918414560
// MI455X (gfx1250) — matrix operand order repaired and checked by reading, not compiled, not run on hardware
//
#include <hip/hip_runtime.h>
#include <stdint.h>

// ---------------------------------------------------------------------------
// Fused GRU-variant scan for MI455X (gfx1250, wave32, WMMA bf16 16x16x32).
//
//   gates(t) = [x_t | h] (16x384) x Wc (384x768),  Wc = [W_ih^T ; W_hh^T]
//   r,z = sigmoid; n = tanh(r * n_pre); h = (1-z)h + z n
//
// 16 WGs (one per 16-row batch tile) x 16 waves. Wave w owns the r/z/n tiles
// for hidden cols j in [16w,16w+16): elementwise update is register-only.
// Weight residency (VGPR cap ~256/wave at 4 waves/SIMD, file ~1024/SIMD):
//   r,z gates ksteps 2..11 : registers (160 VGPRs/wave, whole scan)
//   n gate (12) + r,z ksteps 0..1 (4) : 16 LDS fragment slots (256 KB),
//     packed in fragment order -> two conflict-free ds_load_b128 each.
// A-panel [x_t | h] bf16 is double-buffered -> ONE barrier per timestep.
// Dynamic LDS: 2*12544 + 16*16384 = 287232 B (<= 320 KB/WG per ISA).
// ---------------------------------------------------------------------------

typedef __attribute__((ext_vector_type(16))) __bf16 v16bf;
typedef __attribute__((ext_vector_type(8)))  float  v8f;

#define B_       256
#define T_       1024
#define I_       128
#define H_       256
#define KSTEPS_  12            // 384 / 32
#define LDSKS_   2             // r/z ksteps served from LDS
#define WAVES_   16
#define THREADS_ 512           // 16 waves * 32 lanes

#define A_STRIDE 392           // bf16/row: 384 + 8 pad -> 784 B (16B-aligned rows)
#define APAN_    (16 * A_STRIDE)                       // elems per panel
#define NSLOTS_  16                                    // 12 n + 2 r + 2 z
#define SMEM_BYTES (2 * APAN_ * 2 + NSLOTS_ * WAVES_ * 32 * 32)  // 287232

__global__ __launch_bounds__(THREADS_)
void gru_fused_wmma(const float* __restrict__ x,     // [B, T, I]
                    const float* __restrict__ h0,    // [B, H]
                    const float* __restrict__ w_ih,  // [3H, I]
                    const float* __restrict__ w_hh,  // [3H, H]
                    const float* __restrict__ b_ih,  // [3H]
                    const float* __restrict__ b_hh,  // [3H]
                    float* __restrict__ out)         // [B,T,H] ++ [B,H]
{
    extern __shared__ __align__(16) char smem[];
    __bf16* Apan = (__bf16*)smem;                        // [2][APAN_]
    uint4*  WQ   = (uint4*)(smem + 2 * APAN_ * 2);       // 16 frag slots, 256 KB

    const int tid  = threadIdx.x;
    const int wave = tid >> 5;
    const int lane = tid & 31;
    const int b0   = blockIdx.x * 16;        // first batch row of this WG

    const int nloc = lane & 15;              // N within 16-wide tile
    const int jcol = wave * 16 + nloc;       // hidden unit owned by this lane
    const int mrow = (lane < 16) ? 0 : 8;    // C/D layout row offset

    // LDS fragment-slot addressing: slot s, this wave, this lane -> 2 uint4s
    const int fbase = (wave * 32 + lane) * 2;            // within a slot
#define FRAG_IDX(s) ((s) * (WAVES_ * 64) + fbase)

    // ---- one-time: per-lane combined bias (r/z/n for column jcol) --------
    const float biasr = b_ih[jcol]          + b_hh[jcol];
    const float biasz = b_ih[H_ + jcol]     + b_hh[H_ + jcol];
    const float biasn = b_ih[2 * H_ + jcol] + b_hh[2 * H_ + jcol];

    // ---- one-time: weight B-fragments ------------------------------------
    // B 32x16 bf16 layout (dense analogue of ISA sec 7.12.4 sparse 64x16 B):
    //   lanes 0-15 : N = lane,    K = 0..7, 16..23
    //   lanes16-31 : N = lane-16, K = 8..15, 24..31
    v16bf bfragR[KSTEPS_ - LDSKS_], bfragZ[KSTEPS_ - LDSKS_];  // ksteps 2..11
    {
        const int khalf = (lane < 16) ? 0 : 8;
#pragma unroll
        for (int g = 0; g < 3; ++g) {
            const int n = g * H_ + jcol;           // W row (gate output idx)
#pragma unroll
            for (int ks = 0; ks < KSTEPS_; ++ks) {
                union { v16bf v; __bf16 e[16]; uint4 q[2]; } u;
#pragma unroll
                for (int vv = 0; vv < 8; ++vv) {
                    const int k0 = ks * 32 + khalf + 2 * vv + (vv >= 4 ? 8 : 0);
                    const int k1 = k0 + 1;
                    const float w0 = (k0 < I_) ? w_ih[n * I_ + k0]
                                               : w_hh[n * H_ + (k0 - I_)];
                    const float w1 = (k1 < I_) ? w_ih[n * I_ + k1]
                                               : w_hh[n * H_ + (k1 - I_)];
                    u.e[2 * vv]     = (__bf16)w0;
                    u.e[2 * vv + 1] = (__bf16)w1;
                }
                if (g == 2) {                       // n gate -> LDS slot ks
                    const int fi = FRAG_IDX(ks);
                    WQ[fi] = u.q[0]; WQ[fi + 1] = u.q[1];
                } else if (ks < LDSKS_) {           // r/z ks 0..1 -> slots 12..15
                    const int fi = FRAG_IDX(12 + 2 * ks + g);
                    WQ[fi] = u.q[0]; WQ[fi + 1] = u.q[1];
                } else if (g == 0) bfragR[ks - LDSKS_] = u.v;
                else               bfragZ[ks - LDSKS_] = u.v;
            }
        }
    }

    // ---- one-time: h0 into registers (C/D layout) + panel 0 --------------
    float hreg[8];
#pragma unroll
    for (int r = 0; r < 8; ++r) {
        const int m = r + mrow;
        const float v = h0[(b0 + m) * H_ + jcol];
        hreg[r] = v;
        Apan[m * A_STRIDE + I_ + jcol] = (__bf16)v;
    }
    // stage x_0 into panel 0 (float2 loads, packed bf16 stores)
#pragma unroll
    for (int i = 0; i < 2; ++i) {
        const int idx2 = tid + i * THREADS_;       // 1024 float2 pairs
        const int m = idx2 >> 6, p = idx2 & 63;
        const float2 xv = *(const float2*)&x[((size_t)(b0 + m) * T_) * I_ + 2 * p];
        union { __bf16 e[2]; uint32_t u; } pk;
        pk.e[0] = (__bf16)xv.x; pk.e[1] = (__bf16)xv.y;
        *(uint32_t*)(Apan + m * A_STRIDE + 2 * p) = pk.u;
    }

    // ======================= sequential scan over T =======================
    for (int t = 0; t < T_; ++t) {
        __syncthreads();   // panel (t&1) complete; panel ((t+1)&1) free
        const __bf16* cur = Apan + (t & 1) * APAN_;
        __bf16*       nxt = Apan + ((t + 1) & 1) * APAN_;

        // -- stage x_{t+1} into the next panel (overlaps this step's GEMM) -
        if (t + 1 < T_) {
#pragma unroll
            for (int i = 0; i < 2; ++i) {
                const int idx2 = tid + i * THREADS_;
                const int m = idx2 >> 6, p = idx2 & 63;
                const float2 xv = *(const float2*)
                    &x[((size_t)(b0 + m) * T_ + (t + 1)) * I_ + 2 * p];
                union { __bf16 e[2]; uint32_t u; } pk;
                pk.e[0] = (__bf16)xv.x; pk.e[1] = (__bf16)xv.y;
                *(uint32_t*)(nxt + m * A_STRIDE + 2 * p) = pk.u;
            }
        }

        // -- GEMM: this wave's r/z/n gate tiles over K = 384 ---------------
        v8f accR = {}, accZ = {}, accN = {};
        {
            const int am    = lane & 15;
            const int khalf = (lane < 16) ? 0 : 8;
            const __bf16* arow = cur + am * A_STRIDE;
#pragma unroll
            for (int ks = 0; ks < KSTEPS_; ++ks) {
                // A 16x32 bf16 fragment (ISA 7.12.2): V0-3 = K khalf+{0..7},
                // V4-7 = K 16+khalf+{0..7} -> two aligned 16B LDS loads.
                union { v16bf v; uint4 q[2]; } a;
                a.q[0] = *(const uint4*)(arow + ks * 32 + khalf);
                a.q[1] = *(const uint4*)(arow + ks * 32 + 16 + khalf);
                // n-gate B fragment from LDS slot ks
                union { v16bf v; uint4 q[2]; } wn;
                { const int fi = FRAG_IDX(ks);
                  wn.q[0] = WQ[fi]; wn.q[1] = WQ[fi + 1]; }

                v16bf wr, wz;
                if (ks < LDSKS_) {   // folded at compile time (unrolled loop)
                    union { v16bf v; uint4 q[2]; } tr, tz;
                    { const int fi = FRAG_IDX(12 + 2 * ks + 0);
                      tr.q[0] = WQ[fi]; tr.q[1] = WQ[fi + 1]; }
                    { const int fi = FRAG_IDX(12 + 2 * ks + 1);
                      tz.q[0] = WQ[fi]; tz.q[1] = WQ[fi + 1]; }
                    wr = tr.v; wz = tz.v;
                } else {
                    wr = bfragR[ks - LDSKS_]; wz = bfragZ[ks - LDSKS_];
                }

                accR = __builtin_amdgcn_wmma_f32_16x16x32_bf16(
                           false, a.v, false, wr, (short)0, accR, false, false);
                accZ = __builtin_amdgcn_wmma_f32_16x16x32_bf16(
                           false, a.v, false, wz, (short)0, accZ, false, false);
                accN = __builtin_amdgcn_wmma_f32_16x16x32_bf16(
                           false, a.v, false, wn.v, (short)0, accN, false, false);
            }
        }

        // -- elementwise GRU-variant update: fully in registers ------------
#pragma unroll
        for (int r = 0; r < 8; ++r) {
            const int m = r + mrow;
            const float rp = accR[r] + biasr;
            const float zp = accZ[r] + biasz;
            const float np = accN[r] + biasn;
            const float rs = 1.0f / (1.0f + __expf(-rp));
            const float zs = 1.0f / (1.0f + __expf(-zp));
            const float nn = tanhf(rs * np);   // reset gate scales FULL n pre-act
            const float h  = (1.0f - zs) * hreg[r] + zs * nn;
            hreg[r] = h;
            nxt[m * A_STRIDE + I_ + jcol] = (__bf16)h;          // next-step A
            out[((size_t)(b0 + m) * T_ + t) * H_ + jcol] = h;   // [B,T,H]
        }
    }

    // ---- h_last: appended after [B,T,H] in d_out -------------------------
    const size_t hlast_base = (size_t)B_ * T_ * H_;
#pragma unroll
    for (int r = 0; r < 8; ++r) {
        const int m = r + mrow;
        out[hlast_base + (size_t)(b0 + m) * H_ + jcol] = hreg[r];
    }
}

extern "C" void kernel_launch(void* const* d_in, const int* in_sizes, int n_in,
                              void* d_out, int out_size, void* d_ws, size_t ws_size,
                              hipStream_t stream) {
    (void)in_sizes; (void)n_in; (void)out_size; (void)d_ws; (void)ws_size;
    const float* x    = (const float*)d_in[0];
    const float* h0   = (const float*)d_in[1];
    const float* w_ih = (const float*)d_in[2];
    const float* w_hh = (const float*)d_in[3];
    const float* b_ih = (const float*)d_in[4];
    const float* b_hh = (const float*)d_in[5];
    float* out = (float*)d_out;

    dim3 grid(B_ / 16);        // 16 workgroups, one per 16-row batch tile
    dim3 block(THREADS_);      // 16 wave32 waves
    hipLaunchKernelGGL(gru_fused_wmma, grid, block, SMEM_BYTES, stream,
                       x, h0, w_ih, w_hh, b_ih, b_hh, out);
}
